// HierarchicalGenerator_26809185861757
// MI455X (gfx1250) — matrix operand order repaired and checked by reading, not compiled, not run on hardware
//
#include <hip/hip_runtime.h>
#include <math.h>

#define B_   512
#define T_   128
#define H_   512
#define ZS_  128
#define ZT_  64
#define G3H  1536
#define NIRR 2

typedef __attribute__((ext_vector_type(16))) __bf16 v16bf;
typedef __attribute__((ext_vector_type(8)))  float  v8f;

union V16 { v16bf v; unsigned short u[16]; };

__device__ __forceinline__ unsigned short f2bf(float f) {
    unsigned int x = __float_as_uint(f);
    x += 0x7FFFu + ((x >> 16) & 1u);   // round-to-nearest-even
    return (unsigned short)(x >> 16);
}
__device__ __forceinline__ float sigm(float x) { return 1.f / (1.f + __expf(-x)); }
__device__ __forceinline__ float gumb(float u) { return -__logf(-__logf(u + 1e-10f) + 1e-10f); }

// Async 16-byte global -> LDS copy (gfx1250, tracked by ASYNCcnt).
// Low 32 bits of a generic shared pointer == LDS byte offset.
__device__ __forceinline__ void async_ld16(void* lds_dst, const void* gsrc) {
    unsigned loff = (unsigned)(size_t)lds_dst;
    asm volatile("global_load_async_to_lds_b128 %0, %1, off"
                 : : "v"(loff), "v"(gsrc) : "memory");
}
__device__ __forceinline__ void wait_async0() {
    asm volatile("s_wait_asynccnt 0x0" : : : "memory");
}

// ---------------------------------------------------------------------------
// Operand packing into native gfx1250 WMMA fragment layouts (bf16).
// A fragment (16x32, MxK): lane<16 row=lane&15 K {k0..k0+7, k0+16..k0+23};
//                          lane>=16 same row, K shifted +8.
// B fragment (32x16, KxN): lane = column (lane&15); lanes 0-15 K {k0..k0+7, k0+16..k0+23};
//                          lanes 16-31 K shifted +8.
// Fragments stored contiguously: [tile][kt][lane] * 16 bf16 (1 KB per block).
// ---------------------------------------------------------------------------
__global__ void pack_b_kernel(const float* __restrict__ W, int ldb,
                              unsigned short* __restrict__ Bp, int KT, int n)
{
    int tid = blockIdx.x * blockDim.x + threadIdx.x;
    if (tid >= n) return;
    int lane  = tid & 31;
    int kt    = (tid >> 5) % KT;
    int ntile = tid / (32 * KT);
    int k0  = kt * 32 + (lane >> 4) * 8;
    int col = ntile * 16 + (lane & 15);
    V16 v;
    #pragma unroll
    for (int i = 0; i < 16; ++i) v.u[i] = f2bf(W[(size_t)(k0 + i + (i >> 3) * 8) * ldb + col]);
    ((v16bf*)Bp)[tid] = v.v;
}

__global__ void pack_a_kernel(const float* __restrict__ A, int lda,
                              unsigned short* __restrict__ Ap, int KT, int n)
{
    int tid = blockIdx.x * blockDim.x + threadIdx.x;
    if (tid >= n) return;
    int lane  = tid & 31;
    int kt    = (tid >> 5) % KT;
    int mtile = tid / (32 * KT);
    int row  = mtile * 16 + (lane & 15);
    int koff = (lane >> 4) * 8;
    const float* ap = A + (size_t)row * lda + kt * 32 + koff;
    V16 v;
    #pragma unroll
    for (int i = 0; i < 8; ++i) v.u[i]     = f2bf(ap[i]);
    #pragma unroll
    for (int i = 0; i < 8; ++i) v.u[8 + i] = f2bf(ap[16 + i]);
    ((v16bf*)Ap)[tid] = v.v;
}

// ---------------------------------------------------------------------------
// bf16 WMMA GEMM on pre-packed fragments with async-LDS double buffering.
// Block = 8 waves: 4 in M x 2 in N; block tile 64M x 128N; wave tile 16x64.
// Each stage copies TWO K-slices (24 KB: A 8 KB + B 16 KB) with
// global_load_async_to_lds_b128 while the previous pair is consumed.
// All fragment ds_loads for a pair are hoisted ahead of an 8-WMMA burst.
// ---------------------------------------------------------------------------
__global__ __launch_bounds__(256)
void gemm_pk_wmma(const unsigned short* __restrict__ Ap,
                  const unsigned short* __restrict__ Bp,
                  const float* __restrict__ bias,
                  float* __restrict__ C, int ldc,
                  int KT, int act)
{
    // [buf=2][slice=2][A 2048 ush | B 4096 ush]  -> 48 KB total
    __shared__ __align__(16) unsigned short sh[2 * 2 * 6144];

    const int tid  = threadIdx.x;
    const int lane = tid & 31;
    const int wid  = tid >> 5;
    const int l16  = lane & 15;
    const int hi   = lane >> 4;
    const int mt0  = blockIdx.y * 4;   // first 16-row tile of block
    const int nt0  = blockIdx.x * 8;   // first 16-col tile of block
    const int KP   = KT >> 1;          // K-slice pairs

    auto stage = [&](int kp, int buf) {
        #pragma unroll
        for (int s = 0; s < 2; ++s) {
            const int kt = kp * 2 + s;
            unsigned short* dstA = sh + (buf * 2 + s) * 6144;
            unsigned short* dstB = dstA + 2048;
            {   // A: 256 chunks of 16 B, one per thread
                int ti = tid >> 6, ch = tid & 63;
                const unsigned short* g = Ap + ((size_t)(mt0 + ti) * KT + kt) * 512 + ch * 8;
                async_ld16(dstA + tid * 8, g);
            }
            #pragma unroll
            for (int r = 0; r < 2; ++r) {  // B: 512 chunks, two per thread
                int c  = tid + r * 256;
                int ti = c >> 6, ch = c & 63;
                const unsigned short* g = Bp + ((size_t)(nt0 + ti) * KT + kt) * 512 + ch * 8;
                async_ld16(dstB + c * 8, g);
            }
        }
    };

    v8f acc[4] = {};

    stage(0, 0);
    wait_async0();
    __syncthreads();

    for (int kp = 0; kp < KP; ++kp) {
        const int cur = kp & 1;
        if (kp + 1 < KP) stage(kp + 1, cur ^ 1);   // prefetch next slice pair

        // hoist ALL fragment loads for this pair, then 8 back-to-back WMMAs
        v16bf a[2], b[2][4];
        #pragma unroll
        for (int s = 0; s < 2; ++s) {
            const unsigned short* bufA = sh + (cur * 2 + s) * 6144;
            const unsigned short* bufB = bufA + 2048;
            a[s] = *(const v16bf*)(bufA + ((wid & 3) * 32 + lane) * 16);
            #pragma unroll
            for (int nt = 0; nt < 4; ++nt)
                b[s][nt] = *(const v16bf*)(bufB + (((wid >> 2) * 4 + nt) * 32 + lane) * 16);
        }
        #pragma unroll
        for (int s = 0; s < 2; ++s)
            #pragma unroll
            for (int nt = 0; nt < 4; ++nt)
                acc[nt] = __builtin_amdgcn_wmma_f32_16x16x32_bf16(
                    false, a[s], false, b[s][nt], (short)0, acc[nt], false, false);

        if (kp + 1 < KP) {
            wait_async0();
            __syncthreads();
        }
    }

    #pragma unroll
    for (int nt = 0; nt < 4; ++nt) {
        const int col = (nt0 + (wid >> 2) * 4 + nt) * 16 + l16;
        const float bv = bias ? bias[col] : 0.f;
        #pragma unroll
        for (int r = 0; r < 8; ++r) {
            const int row = (mt0 + (wid & 3)) * 16 + r + (hi << 3);
            float v = acc[nt][r] + bv;
            if (act) v = fmaxf(v, 0.f);
            C[(size_t)row * ldc + col] = v;
        }
    }
}

// ---------------------------------------------------------------------------
__global__ void zero_kernel(float* p, int n) {
    int i = blockIdx.x * blockDim.x + threadIdx.x;
    if (i < n) p[i] = 0.f;
}

// gi0 = base0 + zt_t @ Wih0[0:64] + ts*Wih0[576] + irr0*Wih0[577] + irr1*Wih0[578]
__global__ void gi0_kernel(const float* __restrict__ base0,
                           const float* __restrict__ z_temporal,
                           const float* __restrict__ Wih0,
                           const float* __restrict__ irr,
                           float ts, int t, float* __restrict__ gi0)
{
    int idx = blockIdx.x * blockDim.x + threadIdx.x;
    if (idx >= B_ * G3H) return;
    int b = idx / G3H, n = idx % G3H;
    const float* zt = z_temporal + ((size_t)b * T_ + t) * ZT_;
    float s = base0[idx];
    #pragma unroll 8
    for (int j = 0; j < ZT_; ++j) s += zt[j] * Wih0[(size_t)j * G3H + n];
    s += ts           * Wih0[(size_t)576 * G3H + n];
    s += irr[b*2 + 0] * Wih0[(size_t)577 * G3H + n];
    s += irr[b*2 + 1] * Wih0[(size_t)578 * G3H + n];
    gi0[idx] = s;
}

// PyTorch GRU gate fusion (r,z,n order); gi/gh include biases; in-place h update.
__global__ void gru_combine(const float* __restrict__ gi,
                            const float* __restrict__ gh,
                            float* __restrict__ h)
{
    int idx = blockIdx.x * blockDim.x + threadIdx.x;
    if (idx >= B_ * H_) return;
    int b = idx / H_, i = idx % H_;
    const float* gib = gi + (size_t)b * G3H;
    const float* ghb = gh + (size_t)b * G3H;
    float r = sigm(gib[i]          + ghb[i]);
    float z = sigm(gib[H_ + i]     + ghb[H_ + i]);
    float n = tanhf(gib[2*H_ + i] + r * ghb[2*H_ + i]);
    h[idx] = (1.f - z) * n + z * h[idx];
}

// static heads: static_cont (16 sigmoid), static_cat (gumbel 8/5/12), irr0 (2x binary)
__global__ void static_heads(const float* __restrict__ baseline,
    const float* W_sc, const float* b_sc,
    const float* W_cat0, const float* b_cat0,
    const float* W_cat1, const float* b_cat1,
    const float* W_cat2, const float* b_cat2,
    const float* W_irr, const float* b_irr,
    const float* u_sc0, const float* u_sc1, const float* u_sc2, const float* u_irr,
    float* out_sc, float* out_scat, float* irr)
{
    int tid = blockIdx.x * blockDim.x + threadIdx.x;
    if (tid < B_ * 16) {
        int b = tid / 16, c = tid % 16;
        const float* x = baseline + (size_t)b * H_;
        float s = b_sc[c];
        for (int i = 0; i < H_; ++i) s += x[i] * W_sc[(size_t)i * 16 + c];
        out_sc[tid] = sigm(s);
        return;
    }
    int t2 = tid - B_ * 16;
    if (t2 < B_ * 3) {
        int b = t2 / 3, hsel = t2 % 3;
        const float *Wc, *bc, *u; int w, off;
        if      (hsel == 0) { Wc = W_cat0; bc = b_cat0; u = u_sc0 + b*8;  w = 8;  off = 0;  }
        else if (hsel == 1) { Wc = W_cat1; bc = b_cat1; u = u_sc1 + b*5;  w = 5;  off = 8;  }
        else                { Wc = W_cat2; bc = b_cat2; u = u_sc2 + b*12; w = 12; off = 13; }
        const float* x = baseline + (size_t)b * H_;
        float l[12], m = -1e30f;
        for (int c = 0; c < w; ++c) {
            float s = bc[c];
            for (int i = 0; i < H_; ++i) s += x[i] * Wc[(size_t)i * w + c];
            s += gumb(u[c]);
            l[c] = s; m = fmaxf(m, s);
        }
        float den = 0.f;
        for (int c = 0; c < w; ++c) { l[c] = __expf(l[c] - m); den += l[c]; }
        for (int c = 0; c < w; ++c) out_scat[b * 25 + off + c] = l[c] / den;
        return;
    }
    int t3 = t2 - B_ * 3;
    if (t3 < B_ * NIRR) {
        int b = t3 / NIRR, j = t3 % NIRR;
        const float* x = baseline + (size_t)b * H_;
        float l0 = b_irr[j*2 + 0], l1 = b_irr[j*2 + 1];
        for (int i = 0; i < H_; ++i) {
            float xi = x[i];
            l0 += xi * W_irr[(size_t)j * H_ * 2 + i * 2 + 0];
            l1 += xi * W_irr[(size_t)j * H_ * 2 + i * 2 + 1];
        }
        l0 += gumb(u_irr[(size_t)j * B_ * 2 + b * 2 + 0]);
        l1 += gumb(u_irr[(size_t)j * B_ * 2 + b * 2 + 1]);
        float m = fmaxf(l0, l1);
        float e0 = __expf(l0 - m), e1 = __expf(l1 - m);
        irr[b * 2 + j] = e1 / (e0 + e1);
    }
}

// per-step heads: tc (24 sigmoid), tcat gumbel heads (6/10/4), hazard->irr flip
__global__ void temporal_heads(const float* __restrict__ h1, const int* __restrict__ lengths,
    const float* W_tc, const float* b_tc,
    const float* W_t0, const float* b_t0,
    const float* W_t2, const float* b_t2,
    const float* W_t4, const float* b_t4,
    const float* W_hz, const float* b_hz,
    const float* u_t0, const float* u_t2, const float* u_t4, const float* u_bern,
    float* irr, float* out_tc, float* out_tcat, int t, float ts)
{
    int tid = blockIdx.x * blockDim.x + threadIdx.x;
    if (tid < B_ * 24) {
        int b = tid / 24, c = tid % 24;
        float mask = (t < lengths[b]) ? 1.f : 0.f;
        const float* x = h1 + (size_t)b * H_;
        float s = b_tc[c];
        for (int i = 0; i < H_; ++i) s += x[i] * W_tc[(size_t)i * 24 + c];
        out_tc[((size_t)b * T_ + t) * 24 + c] = sigm(s) * mask;
        return;
    }
    int t2 = tid - B_ * 24;
    if (t2 < B_ * 3) {
        int b = t2 / 3, hsel = t2 % 3;
        float mask = (t < lengths[b]) ? 1.f : 0.f;
        const float *Wc, *bc, *u; int w, off;
        if      (hsel == 0) { Wc = W_t0; bc = b_t0; u = u_t0 + (size_t)t*B_*6  + b*6;  w = 6;  off = 0;  }
        else if (hsel == 1) { Wc = W_t2; bc = b_t2; u = u_t2 + (size_t)t*B_*10 + b*10; w = 10; off = 7;  }
        else                { Wc = W_t4; bc = b_t4; u = u_t4 + (size_t)t*B_*4  + b*4;  w = 4;  off = 18; }
        const float* x = h1 + (size_t)b * H_;
        float l[10], m = -1e30f;
        for (int c = 0; c < w; ++c) {
            float s = bc[c];
            for (int i = 0; i < H_; ++i) s += x[i] * Wc[(size_t)i * w + c];
            s += gumb(u[c]);
            l[c] = s; m = fmaxf(m, s);
        }
        float den = 0.f;
        for (int c = 0; c < w; ++c) { l[c] = __expf(l[c] - m); den += l[c]; }
        float* o = out_tcat + ((size_t)b * T_ + t) * 22;
        for (int c = 0; c < w; ++c) o[off + c] = (l[c] / den) * mask;
        return;
    }
    int t3 = t2 - B_ * 3;
    if (t3 < B_) {
        int b = t3;
        float mask = (t < lengths[b]) ? 1.f : 0.f;
        const float* x = h1 + (size_t)b * H_;
        float* o = out_tcat + ((size_t)b * T_ + t) * 22;
        #pragma unroll
        for (int j = 0; j < NIRR; ++j) {
            const float* wz = W_hz + (size_t)j * (H_ + 1);
            float s = b_hz[j];
            for (int i = 0; i < H_; ++i) s += x[i] * wz[i];
            s += ts * wz[H_];
            float haz = sigm(s);
            float ir  = irr[b * 2 + j];
            float ub  = u_bern[(size_t)t * B_ * 2 + b * 2 + j];
            float flip = (ub < haz * (1.f - ir)) ? 1.f : 0.f;
            float nv = fminf(ir + flip, 1.f);
            irr[b * 2 + j] = nv;
            o[j == 0 ? 6 : 17] = nv * mask;
        }
    }
}

__global__ void mask_times(const int* __restrict__ lengths, float* out_mask, float* out_times) {
    int idx = blockIdx.x * blockDim.x + threadIdx.x;
    if (idx >= B_ * T_) return;
    int b = idx / T_, t = idx % T_;
    out_mask[idx]  = (t < lengths[b]) ? 1.f : 0.f;
    out_times[idx] = (float)t / (float)(T_ - 1);
}

// ---------------------------------------------------------------------------
extern "C" void kernel_launch(void* const* d_in, const int* in_sizes, int n_in,
                              void* d_out, int out_size, void* d_ws, size_t ws_size,
                              hipStream_t stream)
{
    const float* z_static   = (const float*)d_in[0];
    const float* z_temporal = (const float*)d_in[1];
    const int*   lengths    = (const int*)  d_in[2];
    const float* Ws1  = (const float*)d_in[3];  const float* bs1  = (const float*)d_in[4];
    const float* Ws2  = (const float*)d_in[5];  const float* bs2  = (const float*)d_in[6];
    const float* W_sc = (const float*)d_in[7];  const float* b_sc = (const float*)d_in[8];
    const float* W_c0 = (const float*)d_in[9];  const float* b_c0 = (const float*)d_in[10];
    const float* W_c1 = (const float*)d_in[11]; const float* b_c1 = (const float*)d_in[12];
    const float* W_c2 = (const float*)d_in[13]; const float* b_c2 = (const float*)d_in[14];
    const float* W_irr = (const float*)d_in[15]; const float* b_irr = (const float*)d_in[16];
    const float* Wih0 = (const float*)d_in[17]; const float* Whh0 = (const float*)d_in[18];
    const float* bih0 = (const float*)d_in[19]; const float* bhh0 = (const float*)d_in[20];
    const float* Wih1 = (const float*)d_in[21]; const float* Whh1 = (const float*)d_in[22];
    const float* bih1 = (const float*)d_in[23]; const float* bhh1 = (const float*)d_in[24];
    const float* W_tc = (const float*)d_in[25]; const float* b_tc = (const float*)d_in[26];
    const float* W_t0 = (const float*)d_in[27]; const float* b_t0 = (const float*)d_in[28];
    const float* W_t2 = (const float*)d_in[29]; const float* b_t2 = (const float*)d_in[30];
    const float* W_t4 = (const float*)d_in[31]; const float* b_t4 = (const float*)d_in[32];
    const float* W_hz = (const float*)d_in[33]; const float* b_hz = (const float*)d_in[34];
    const float* u_s0 = (const float*)d_in[35]; const float* u_s1 = (const float*)d_in[36];
    const float* u_s2 = (const float*)d_in[37]; const float* u_ir = (const float*)d_in[38];
    const float* u_t0 = (const float*)d_in[39]; const float* u_t2 = (const float*)d_in[40];
    const float* u_t4 = (const float*)d_in[41]; const float* u_bn = (const float*)d_in[42];

    // ---- workspace carve-out: f32 region ----
    float* ws = (float*)d_ws;
    float* tmp1     = ws; ws += B_ * H_;
    float* baseline = ws; ws += B_ * H_;
    float* base0    = ws; ws += B_ * G3H;
    float* h0       = ws; ws += B_ * H_;
    float* h1       = ws; ws += B_ * H_;   // h0,h1 contiguous -> single zero pass
    float* gi0      = ws; ws += B_ * G3H;
    float* gh0      = ws; ws += B_ * G3H;
    float* gi1      = ws; ws += B_ * G3H;
    float* gh1      = ws; ws += B_ * G3H;
    float* irr      = ws; ws += B_ * NIRR;

    // ---- workspace carve-out: packed bf16 fragment region ----
    unsigned short* wp = (unsigned short*)ws;
    unsigned short* h0_p    = wp; wp += B_ * H_;
    unsigned short* h1_p    = wp; wp += B_ * H_;   // adjacent: zeroed together
    unsigned short* zs_p    = wp; wp += B_ * ZS_;
    unsigned short* tmp1_p  = wp; wp += B_ * H_;
    unsigned short* base_p  = wp; wp += B_ * H_;
    unsigned short* Ws1_p   = wp; wp += ZS_ * H_;
    unsigned short* Ws2_p   = wp; wp += H_ * H_;
    unsigned short* Wih0s_p = wp; wp += H_ * G3H;
    unsigned short* Whh0_p  = wp; wp += H_ * G3H;
    unsigned short* Wih1_p  = wp; wp += H_ * G3H;
    unsigned short* Whh1_p  = wp; wp += H_ * G3H;

    // ---- output carve-out (return order, flat) ----
    float* out     = (float*)d_out;
    float* o_sc    = out;            // [B,16]
    float* o_scat  = out + 8192;     // [B,25]
    float* o_tc    = out + 20992;    // [B,T,24]
    float* o_tcat  = out + 1593856;  // [B,T,22]
    float* o_mask  = out + 3035648;  // [B,T,1]
    float* o_times = out + 3101184;  // [B,T]

    const dim3 blk(256);
    auto packB = [&](const float* W, int ldb, unsigned short* dst, int N, int K) {
        int KT = K / 32, n = (N / 16) * KT * 32;
        pack_b_kernel<<<(n + 255) / 256, blk, 0, stream>>>(W, ldb, dst, KT, n);
    };
    auto packA = [&](const float* A, int lda, unsigned short* dst, int M, int K) {
        int KT = K / 32, n = (M / 16) * KT * 32;
        pack_a_kernel<<<(n + 255) / 256, blk, 0, stream>>>(A, lda, dst, KT, n);
    };
    auto gemm = [&](const unsigned short* Apk, const unsigned short* Bpk, const float* bias,
                    float* C, int M, int N, int K, int act) {
        dim3 grid(N / 128, M / 64);
        gemm_pk_wmma<<<grid, blk, 0, stream>>>(Apk, Bpk, bias, C, N, K / 32, act);
    };

    // one-time weight fragment packing (L2-resident for the whole scan)
    packB(Ws1, H_, Ws1_p, H_, ZS_);
    packB(Ws2, H_, Ws2_p, H_, H_);
    packB(Wih0 + (size_t)64 * G3H, G3H, Wih0s_p, G3H, H_);
    packB(Whh0, G3H, Whh0_p, G3H, H_);
    packB(Wih1, G3H, Wih1_p, G3H, H_);
    packB(Whh1, G3H, Whh1_p, G3H, H_);
    packA(z_static, ZS_, zs_p, B_, ZS_);

    // zero f32 h-state and the packed h fragments (bf16 zero == 0x0000)
    zero_kernel<<<(2 * B_ * H_ + 255) / 256, blk, 0, stream>>>(h0, 2 * B_ * H_);
    zero_kernel<<<(B_ * H_ + 255) / 256, blk, 0, stream>>>((float*)h0_p, B_ * H_);

    // static trunk
    gemm(zs_p, Ws1_p, bs1, tmp1, B_, H_, ZS_, 1);
    packA(tmp1, H_, tmp1_p, B_, H_);
    gemm(tmp1_p, Ws2_p, bs2, baseline, B_, H_, H_, 1);
    packA(baseline, H_, base_p, B_, H_);
    gemm(base_p, Wih0s_p, bih0, base0, B_, G3H, H_, 0);   // loop-invariant input proj

    static_heads<<<(B_ * 16 + B_ * 3 + B_ * NIRR + 255) / 256, blk, 0, stream>>>(
        baseline, W_sc, b_sc, W_c0, b_c0, W_c1, b_c1, W_c2, b_c2,
        W_irr, b_irr, u_s0, u_s1, u_s2, u_ir, o_sc, o_scat, irr);
    mask_times<<<(B_ * T_ + 255) / 256, blk, 0, stream>>>(lengths, o_mask, o_times);

    for (int t = 0; t < T_; ++t) {
        const float ts = (float)t / (float)(T_ - 1);
        gi0_kernel<<<(B_ * G3H + 255) / 256, blk, 0, stream>>>(
            base0, z_temporal, Wih0, irr, ts, t, gi0);
        gemm(h0_p, Whh0_p, bhh0, gh0, B_, G3H, H_, 0);         // uses h0 from t-1
        gru_combine<<<(B_ * H_ + 255) / 256, blk, 0, stream>>>(gi0, gh0, h0);
        packA(h0, H_, h0_p, B_, H_);
        gemm(h0_p, Wih1_p, bih1, gi1, B_, G3H, H_, 0);         // uses fresh h0
        gemm(h1_p, Whh1_p, bhh1, gh1, B_, G3H, H_, 0);         // uses h1 from t-1
        gru_combine<<<(B_ * H_ + 255) / 256, blk, 0, stream>>>(gi1, gh1, h1);
        packA(h1, H_, h1_p, B_, H_);
        temporal_heads<<<(B_ * 24 + B_ * 3 + B_ + 255) / 256, blk, 0, stream>>>(
            h1, lengths, W_tc, b_tc, W_t0, b_t0, W_t2, b_t2, W_t4, b_t4,
            W_hz, b_hz, u_t0, u_t2, u_t4, u_bn, irr, o_tc, o_tcat, t, ts);
    }
}
